// Model_71966472012276
// MI455X (gfx1250) — matrix operand order repaired and checked by reading, not compiled, not run on hardware
//
#include <hip/hip_runtime.h>
#include <hip/hip_bf16.h>

// ---------------------------------------------------------------------------
// MI455X (gfx1250) grouped per-species MLP, bf16 WMMA with f32 accumulate.
//   h1 = silu(X @ W1 + b1)   X:[4x4096x2880] fp32, W1:[4x2880x256]
//   h2 = silu(h1 @ W2 + b2)  W2:[4x256x256]
//   e  = h2 . w_last + b_last + comp[s]; out = segment_sum(e, structure_idx)
//
// Roofline: features (189MB fp32) are the HBM floor (~8.5us @ 23.3TB/s);
// bf16 WMMA puts the 26 GFLOP of matrix math at the same scale. Weights are
// pre-swizzled once per launch into WMMA B-fragment order so the hot loop
// stages them with GLOBAL_LOAD_ASYNC_TO_LDS_B128 (pure async copy, no VALU)
// and reads fragments as contiguous ds_load_b128.
// ---------------------------------------------------------------------------

typedef __attribute__((ext_vector_type(16))) __bf16 v16bf;
typedef __attribute__((ext_vector_type(8)))  __bf16 v8bf;
typedef __attribute__((ext_vector_type(4)))  __bf16 v4bf;
typedef __attribute__((ext_vector_type(2)))  __bf16 v2bf;
typedef __attribute__((ext_vector_type(8)))  float  v8f;
typedef __attribute__((ext_vector_type(4)))  float  v4f;
typedef __attribute__((ext_vector_type(2)))  float  v2f;

#define D_IN    2880
#define HDIM    256
#define NSPEC   4
#define NPER    4096
#define NATOMS  16384
#define NSTRUCT 64

__device__ __forceinline__ float silu_f(float x) {
    return x / (1.0f + __expf(-x));
}

// --- CDNA5 async global->LDS copy (ASYNCcnt path), 16B per lane -------------
__device__ __forceinline__ void async_copy_b128(unsigned lds_byte_addr,
                                                const void* gbase,
                                                int byte_off) {
    asm volatile("global_load_async_to_lds_b128 %0, %1, %2"
                 :: "v"(lds_byte_addr), "v"(byte_off), "s"(gbase)
                 : "memory");
}
__device__ __forceinline__ void wait_async0() {
#if __has_builtin(__builtin_amdgcn_s_wait_asynccnt)
    __builtin_amdgcn_s_wait_asynccnt(0);
#else
    asm volatile("s_wait_asynccnt 0" ::: "memory");
#endif
}
__device__ __forceinline__ unsigned lds_addr_of(const void* p) {
    // LDS aperture: flat addr[31:0] is the LDS byte address (ISA 10.2).
    return (unsigned)(unsigned long long)p;
}

// A-fragment (v_wmma_f32_16x16x32_bf16) from a row-major [row][32] bf16 tile:
// lanes 0-15 hold K={0..15}, lanes 16-31 K={16..31}; element e is
// K = e + 16*khalf -> two contiguous 16B chunks per lane.
__device__ __forceinline__ v16bf a_frag_from(const __bf16* rowp, int khalf) {
    v8bf lo = *(const v8bf*)(rowp + khalf * 16);
    v8bf hi = *(const v8bf*)(rowp + 8 + khalf * 16);
    return __builtin_shufflevector(lo, hi, 0,1,2,3,4,5,6,7,8,9,10,11,12,13,14,15);
}

// ---------------------------------------------------------------------------
// Kernel 0: swizzle fp32 weights [S][Kdim][256] into bf16 B-fragment order:
// per species, per (k-chunk c, n-tile j) a 1KB block laid out so lane l owns
// 32 contiguous bytes: n = 16j + (l&15), K = 32c + 16*(l>>4) + e, e = 0..15.
// One thread per output dword. One-time cost (~14MB read), L2-resident after.
// ---------------------------------------------------------------------------
__global__ __launch_bounds__(256) void swizzle_w_kernel(
    const float* __restrict__ w, __bf16* __restrict__ wf, int Kdim)
{
    const int s = blockIdx.y;
    const int g = blockIdx.x * 256 + threadIdx.x;   // dword id within species
    const int blk  = g >> 8;                         // 1KB fragment block
    const int win  = g & 255;
    const int lane = win >> 3;
    const int ep   = win & 7;
    const int c = blk >> 4;
    const int j = blk & 15;
    const int K = 32 * c + ((lane >> 4) * 16) + 2 * ep;
    const int n = 16 * j + (lane & 15);
    const float f0 = w[((size_t)s * Kdim + K) * HDIM + n];
    const float f1 = w[((size_t)s * Kdim + K + 1) * HDIM + n];
    v2f ff = { f0, f1 };
    ((v2bf*)wf)[(size_t)s * Kdim * 128 + g] = __builtin_convertvector(ff, v2bf);
}

// ---------------------------------------------------------------------------
// Kernel 1: h1 = silu(X @ W1 + b1). Block tile 64(M) x 256(N), K-chunks of 32.
// 8 waves: wave w -> mg = w>>1 (16 rows), nblk = w&1 (8 N-tiles of 16).
// B chunk (16KB, pre-swizzled bf16) staged via async-to-LDS; A chunk staged
// with vector fp32 loads + packed bf16 converts. Epilogue bounces through LDS
// so h1 is written with coalesced b128 stores, row-major.
// ---------------------------------------------------------------------------
__global__ __launch_bounds__(256) void gemm1_silu_kernel(
    const float* __restrict__ feat, const __bf16* __restrict__ w1f,
    const float* __restrict__ b1, __bf16* __restrict__ h1)
{
    __shared__ __align__(128) unsigned char smem[4096 + 32768];
    __bf16* ldsA   = (__bf16*)smem;            // [64][32] row-major, 4KB
    __bf16* ldsB   = (__bf16*)(smem + 4096);   // fragment-ordered, 16KB
    __bf16* ldsOut = (__bf16*)(smem + 4096);   // epilogue bounce, 32KB (reuse)

    const int tid  = threadIdx.x;
    const int lane = tid & 31;
    const int w    = tid >> 5;
    const int m0   = blockIdx.x * 64;
    const int s    = blockIdx.y;

    const float*  A  = feat + (size_t)(s * NPER + m0) * D_IN;
    const __bf16* Bf = w1f + (size_t)s * D_IN * HDIM;

    v8f acc[8];
    #pragma unroll
    for (int t = 0; t < 8; ++t) acc[t] = (v8f)0.0f;

    const int mg    = w >> 1;
    const int nblk  = w & 1;
    const int rsel  = lane & 15;
    const int khalf = lane >> 4;

    const unsigned ldsB_base = lds_addr_of(ldsB);
    const int ar = tid >> 2;           // A staging: row
    const int ak = (tid & 3) * 8;      // A staging: k offset (8 per thread)

    for (int c = 0; c < D_IN / 32; ++c) {
        // Async-stage B: 16KB pre-swizzled chunk, pure copy, 4x16B per thread.
        const void* gB = (const void*)(Bf + (size_t)c * 8192);
        #pragma unroll
        for (int q = 0; q < 4; ++q) {
            int off = tid * 64 + q * 16;
            async_copy_b128(ldsB_base + off, gB, off);
        }
        // Stage A: 64x32 fp32 -> bf16 (2x b128 loads, packed cvt, 1x b128 store).
        {
            const float* ap = A + (size_t)ar * D_IN + c * 32 + ak;
            v4f f0 = *(const v4f*)(ap);
            v4f f1 = *(const v4f*)(ap + 4);
            v8bf bb;
            *(v4bf*)&bb    = __builtin_convertvector(f0, v4bf);
            *((v4bf*)&bb + 1) = __builtin_convertvector(f1, v4bf);
            *(v8bf*)(ldsA + ar * 32 + ak) = bb;
        }
        wait_async0();
        __syncthreads();

        v16bf afrag = a_frag_from(ldsA + (mg * 16 + rsel) * 32, khalf);
        #pragma unroll
        for (int t = 0; t < 8; ++t) {
            int j = nblk * 8 + t;
            v16bf bfrag = *(const v16bf*)(ldsB + j * 512 + lane * 16);
            acc[t] = __builtin_amdgcn_wmma_f32_16x16x32_bf16(
                false, afrag, false, bfrag, (short)0, acc[t], false, false);
        }
        __syncthreads();
    }

    // Epilogue: bias + SiLU -> LDS scatter (b16), then coalesced b128 stores.
    const int hi = lane >> 4;
    const int nl = lane & 15;
    #pragma unroll
    for (int t = 0; t < 8; ++t) {
        int col = (nblk * 8 + t) * 16 + nl;
        float bias = b1[s * HDIM + col];
        #pragma unroll
        for (int r = 0; r < 8; ++r) {
            int rl = mg * 16 + r + 8 * hi;
            ldsOut[rl * HDIM + col] = (__bf16)silu_f(acc[t][r] + bias);
        }
    }
    __syncthreads();
    {
        __bf16* dst = h1 + (size_t)(s * NPER + m0) * HDIM + tid * 64;
        const __bf16* src = ldsOut + tid * 64;
        #pragma unroll
        for (int q = 0; q < 8; ++q)
            *(v8bf*)(dst + q * 8) = *(const v8bf*)(src + q * 8);
    }
}

// ---------------------------------------------------------------------------
// Kernel 2: h2 = silu(h1 @ W2 + b2); e = h2 . w_last + b_last + comp.
// h1 is row-major bf16, so A-fragments load directly from global (L2-resident,
// 2x contiguous b128 per lane) -- no A staging. B via async-to-LDS as above.
// Deterministic epilogue: butterfly shuffles + fixed-order LDS combine.
// ---------------------------------------------------------------------------
__global__ __launch_bounds__(256) void gemm2_energy_kernel(
    const __bf16* __restrict__ h1, const __bf16* __restrict__ w2f,
    const float* __restrict__ b2, const float* __restrict__ w_last,
    const float* __restrict__ b_last, const float* __restrict__ comp,
    float* __restrict__ e_atom)
{
    __shared__ __align__(128) __bf16 ldsB[HDIM * 32];  // 16KB
    __shared__ float e_part[2][64];

    const int tid  = threadIdx.x;
    const int lane = tid & 31;
    const int w    = tid >> 5;
    const int m0   = blockIdx.x * 64;
    const int s    = blockIdx.y;

    const __bf16* Bf = w2f + (size_t)s * HDIM * HDIM;

    v8f acc[8];
    #pragma unroll
    for (int t = 0; t < 8; ++t) acc[t] = (v8f)0.0f;

    const int mg    = w >> 1;
    const int nblk  = w & 1;
    const int rsel  = lane & 15;
    const int khalf = lane >> 4;
    const unsigned ldsB_base = lds_addr_of(ldsB);

    // This lane's A row in row-major h1.
    const __bf16* Arow = h1 + (size_t)(s * NPER + m0 + mg * 16 + rsel) * HDIM;

    for (int c = 0; c < HDIM / 32; ++c) {
        const void* gB = (const void*)(Bf + (size_t)c * 8192);
        #pragma unroll
        for (int q = 0; q < 4; ++q) {
            int off = tid * 64 + q * 16;
            async_copy_b128(ldsB_base + off, gB, off);
        }
        wait_async0();
        __syncthreads();

        v16bf afrag = a_frag_from(Arow + c * 32, khalf);
        #pragma unroll
        for (int t = 0; t < 8; ++t) {
            int j = nblk * 8 + t;
            v16bf bfrag = *(const v16bf*)(ldsB + j * 512 + lane * 16);
            acc[t] = __builtin_amdgcn_wmma_f32_16x16x32_bf16(
                false, afrag, false, bfrag, (short)0, acc[t], false, false);
        }
        __syncthreads();
    }

    // silu + dot with w_last, reduce across the 16 lanes sharing a row.
    const int hi = lane >> 4;
    const int nl = lane & 15;
    float psum[8];
    #pragma unroll
    for (int r = 0; r < 8; ++r) psum[r] = 0.0f;
    #pragma unroll
    for (int t = 0; t < 8; ++t) {
        int n = (nblk * 8 + t) * 16 + nl;
        float bias = b2[s * HDIM + n];
        float wl   = w_last[s * HDIM + n];
        #pragma unroll
        for (int r = 0; r < 8; ++r)
            psum[r] += silu_f(acc[t][r] + bias) * wl;
    }
    #pragma unroll
    for (int r = 0; r < 8; ++r) {
        float v = psum[r];
        v += __shfl_xor(v, 1, 32);
        v += __shfl_xor(v, 2, 32);
        v += __shfl_xor(v, 4, 32);
        v += __shfl_xor(v, 8, 32);
        if (nl == 0) e_part[nblk][mg * 16 + 8 * hi + r] = v;
    }
    __syncthreads();
    if (tid < 64) {
        int atom = s * NPER + m0 + tid;
        e_atom[atom] = e_part[0][tid] + e_part[1][tid] + b_last[s] + comp[s];
    }
}

// ---------------------------------------------------------------------------
// Kernel 3: deterministic segment sum, one block per structure.
// ---------------------------------------------------------------------------
__global__ __launch_bounds__(256) void segment_sum_kernel(
    const float* __restrict__ e_atom, const int* __restrict__ sidx,
    float* __restrict__ out)
{
    __shared__ float red[256];
    const int b = blockIdx.x, tid = threadIdx.x;
    float p = 0.0f;
    for (int i = tid; i < NATOMS; i += 256)
        p += (sidx[i] == b) ? e_atom[i] : 0.0f;
    red[tid] = p;
    __syncthreads();
    for (int off = 128; off > 0; off >>= 1) {
        if (tid < off) red[tid] += red[tid + off];
        __syncthreads();
    }
    if (tid == 0) out[b] = red[0];
}

// ---------------------------------------------------------------------------
extern "C" void kernel_launch(void* const* d_in, const int* in_sizes, int n_in,
                              void* d_out, int out_size, void* d_ws, size_t ws_size,
                              hipStream_t stream) {
    const float* feat   = (const float*)d_in[0];
    const float* w1     = (const float*)d_in[1];
    const float* b1     = (const float*)d_in[2];
    const float* w2     = (const float*)d_in[3];
    const float* b2     = (const float*)d_in[4];
    const float* w_last = (const float*)d_in[5];
    const float* b_last = (const float*)d_in[6];
    const float* comp   = (const float*)d_in[7];
    const int*   sidx   = (const int*)d_in[8];
    float* out = (float*)d_out;

    // ws: h1 bf16 (8MB) | e_atom (64KB) | w1f bf16 (5.9MB) | w2f bf16 (512KB)
    char* ws = (char*)d_ws;
    __bf16* h1     = (__bf16*)ws;
    float*  e_atom = (float*)(ws + (size_t)NATOMS * HDIM * 2);
    __bf16* w1f    = (__bf16*)(ws + (size_t)NATOMS * HDIM * 2 + 65536);
    __bf16* w2f    = (__bf16*)(ws + (size_t)NATOMS * HDIM * 2 + 65536
                                  + (size_t)NSPEC * D_IN * HDIM * 2);

    // Per-launch weight swizzle to bf16 fragment order (deterministic).
    dim3 gs1(D_IN * 128 / 256, NSPEC);   // 1440 x 4
    swizzle_w_kernel<<<gs1, 256, 0, stream>>>(w1, w1f, D_IN);
    dim3 gs2(HDIM * 128 / 256, NSPEC);   // 128 x 4
    swizzle_w_kernel<<<gs2, 256, 0, stream>>>(w2, w2f, HDIM);

    dim3 g1(NPER / 64, NSPEC);
    gemm1_silu_kernel<<<g1, 256, 0, stream>>>(feat, w1f, b1, h1);

    dim3 g2(NPER / 64, NSPEC);
    gemm2_energy_kernel<<<g2, 256, 0, stream>>>(h1, w2f, b2, w_last, b_last, comp, e_atom);

    segment_sum_kernel<<<NSTRUCT, 256, 0, stream>>>(e_atom, sidx, out);
}
